// GIN_40407052320950
// MI455X (gfx1250) — matrix operand order repaired and checked by reading, not compiled, not run on hardware
//
#include <hip/hip_runtime.h>
#include <hip/hip_bf16.h>

// ---------------------------------------------------------------------------
// GIN (3 layers, H=128) inference for MI455X (gfx1250, wave32, WMMA).
//  - Edge aggregation: f32 global atomics (working set fits 192MB L2).
//  - Layer GEMMs (layers 2,3): v_wmma_f32_16x16x32_bf16, weights pre-packed
//    into WMMA B-fragment layout and staged through LDS.
//  - BN folded to per-channel scale/shift; ReLU fused into the GEMM epilogue.
// Workspace use: ~104.3 MB.
// ---------------------------------------------------------------------------

#define NNODES 100000
#define NEDGES 1600000
#define NGRAPH 256
#define HD     128

typedef __attribute__((ext_vector_type(16))) __bf16 v16bf;
typedef __attribute__((ext_vector_type(8)))  float  v8f;

// ---------------- small prep kernels ----------------

// BN eval:  z*scale + shift  with scale = g/sqrt(v+eps), shift = b - m*scale
__global__ void bn_prep(const float* __restrict__ g, const float* __restrict__ b,
                        const float* __restrict__ m, const float* __restrict__ v,
                        float* __restrict__ scale, float* __restrict__ shift) {
    int i = threadIdx.x;
    if (i < HD) {
        float s = g[i] * rsqrtf(v[i] + 1e-5f);
        scale[i] = s;
        shift[i] = b[i] - m[i] * s;
    }
}

__global__ void zero_f(float* __restrict__ p, int n) {
    int i = blockIdx.x * 256 + threadIdx.x;
    if (i < n) p[i] = 0.0f;
}

__global__ void copy4(const float4* __restrict__ src, float4* __restrict__ dst, int n4) {
    int i = blockIdx.x * 256 + threadIdx.x;
    if (i < n4) dst[i] = src[i];
}

// Pack W[128][128] (row-major [k][o], f32) into bf16 WMMA B-fragment order:
//   Wb[((kc*8 + ot)*32 + lane)*16 + j] = W[(kc*32 + 16*(lane>>4) + j)*128 + ot*16 + (lane&15)]
// lane holds column N = lane%16, K-half = lane/16; VGPR i holds K = 16*half + 2i, 2i+1.
__global__ void pack_w(const float* __restrict__ W, __bf16* __restrict__ Wb) {
    int tid = blockIdx.x * 256 + threadIdx.x;   // 16384 total
    int j    =  tid        & 15;
    int lane = (tid >> 4)  & 31;
    int ot   = (tid >> 9)  & 7;
    int kc   =  tid >> 12;
    int k = kc * 32 + ((lane >> 4) << 4) + j;
    int o = (ot << 4) + (lane & 15);
    Wb[tid] = (__bf16)W[k * HD + o];
}

// ---------------- layer 1 (K=4, scalar) ----------------

__global__ void __launch_bounds__(256)
gin_layer1(const float* __restrict__ S4, const float* __restrict__ W1,
           const float* __restrict__ scale, const float* __restrict__ shift,
           float* __restrict__ out) {
    int gid = blockIdx.x * 256 + threadIdx.x;       // NNODES*HD threads exactly
    int n = gid >> 7;
    int o = gid & (HD - 1);
    float4 s = *(const float4*)(S4 + n * 4);
    float z = s.x * W1[o] + s.y * W1[HD + o] + s.z * W1[2 * HD + o] + s.w * W1[3 * HD + o];
    z = z * scale[o] + shift[o];
    out[gid] = fmaxf(z, 0.0f);
}

// ---------------- edge aggregation ----------------

// dim-4 aggregation for layer 1: one thread per edge
__global__ void edge_aggr4(const float* __restrict__ x, const long long* __restrict__ ei,
                           float* __restrict__ S4) {
    int e = blockIdx.x * 256 + threadIdx.x;
    if (e >= NEDGES) return;
    int s = (int)ei[e];
    int d = (int)ei[NEDGES + e];
    float4 v = *(const float4*)(x + (size_t)s * 4);
    float* p = S4 + (size_t)d * 4;
    atomicAdd(p + 0, v.x); atomicAdd(p + 1, v.y);
    atomicAdd(p + 2, v.z); atomicAdd(p + 3, v.w);
}

// dim-128 aggregation: one wave32 per edge, 4 floats per lane
__global__ void edge_aggr128(const float* __restrict__ h, const long long* __restrict__ ei,
                             float* __restrict__ S) {
    int lane = threadIdx.x & 31;
    int e = blockIdx.x * 8 + (threadIdx.x >> 5);
    if (e >= NEDGES) return;
    int s = (int)ei[e];
    int d = (int)ei[NEDGES + e];
    float4 v = *(const float4*)(h + (size_t)s * HD + lane * 4);
    float* p = S + (size_t)d * HD + lane * 4;
    atomicAdd(p + 0, v.x); atomicAdd(p + 1, v.y);
    atomicAdd(p + 2, v.z); atomicAdd(p + 3, v.w);
}

// ---------------- fused GEMM + BN + ReLU via WMMA bf16 ----------------
// out[n][o] = relu( (sum_k S[n][k]*W[k][o]) * scale[o] + shift[o] )
// One wave handles a 16-row x 128-col stripe: 8 output tiles x 4 K-chunks = 32 WMMAs.

__global__ void __launch_bounds__(256)
gin_gemm_wmma(const float* __restrict__ S, const __bf16* __restrict__ Wb,
              const float* __restrict__ scale, const float* __restrict__ shift,
              float* __restrict__ out) {
    __shared__ uint4 ldsW[2048];                      // 32 KB packed bf16 weights
    const uint4* Wb4 = reinterpret_cast<const uint4*>(Wb);
    int t = threadIdx.x;
#pragma unroll
    for (int i = 0; i < 8; ++i) ldsW[t + 256 * i] = Wb4[t + 256 * i];
    __syncthreads();

    int lane = t & 31;
    int wave = t >> 5;
    int tile = blockIdx.x * 8 + wave;                 // 16-row tile index
    int r0 = tile * 16;
    if (r0 >= NNODES) return;                         // uniform within wave

    int half = lane >> 4;                             // K-half selector
    const float* Srow = S + (size_t)(r0 + (lane & 15)) * HD;

    // A fragments: lane = row (lane%16); j=0..15 -> K=kc*32+half*16+j
    v16bf a[4];
#pragma unroll
    for (int kc = 0; kc < 4; ++kc) {
        int base = kc * 32 + half * 16;
        float4 p0 = *(const float4*)(Srow + base);
        float4 p1 = *(const float4*)(Srow + base + 4);
        float4 p2 = *(const float4*)(Srow + base + 8);
        float4 p3 = *(const float4*)(Srow + base + 12);
        a[kc][0]  = (__bf16)p0.x; a[kc][1]  = (__bf16)p0.y;
        a[kc][2]  = (__bf16)p0.z; a[kc][3]  = (__bf16)p0.w;
        a[kc][4]  = (__bf16)p1.x; a[kc][5]  = (__bf16)p1.y;
        a[kc][6]  = (__bf16)p1.z; a[kc][7]  = (__bf16)p1.w;
        a[kc][8]  = (__bf16)p2.x; a[kc][9]  = (__bf16)p2.y;
        a[kc][10] = (__bf16)p2.z; a[kc][11] = (__bf16)p2.w;
        a[kc][12] = (__bf16)p3.x; a[kc][13] = (__bf16)p3.y;
        a[kc][14] = (__bf16)p3.z; a[kc][15] = (__bf16)p3.w;
    }

#pragma unroll
    for (int ot = 0; ot < 8; ++ot) {
        v8f acc = {};
#pragma unroll
        for (int kc = 0; kc < 4; ++kc) {
            int f = (kc * 8 + ot) * 32 + lane;        // fragment index
            v16bf b = *reinterpret_cast<const v16bf*>(&ldsW[f * 2]);
            acc = __builtin_amdgcn_wmma_f32_16x16x32_bf16(
                false, a[kc], false, b, (short)0, acc, false, false);
        }
        int col = ot * 16 + (lane & 15);              // C layout: N = lane%16
        float sc = scale[col], sh = shift[col];
#pragma unroll
        for (int i = 0; i < 8; ++i) {                 // C layout: M = half*8 + i
            int row = r0 + half * 8 + i;
            float v = fmaxf(acc[i] * sc + sh, 0.0f);
            out[(size_t)row * HD + col] = v;
        }
    }
}

// ---------------- pooling + head ----------------

__global__ void pool_kernel(const float* __restrict__ h, const long long* __restrict__ batch,
                            float* __restrict__ pool, float* __restrict__ cnt) {
    int lane = threadIdx.x & 31;
    int n = blockIdx.x * 8 + (threadIdx.x >> 5);
    if (n >= NNODES) return;
    int g = (int)batch[n];
    float4 v = *(const float4*)(h + (size_t)n * HD + lane * 4);
    float* p = pool + (size_t)g * HD + lane * 4;
    atomicAdd(p + 0, v.x); atomicAdd(p + 1, v.y);
    atomicAdd(p + 2, v.z); atomicAdd(p + 3, v.w);
    if (lane == 0) atomicAdd(cnt + g, 1.0f);
}

__global__ void head_kernel(const float* __restrict__ pool, const float* __restrict__ cnt,
                            const float* __restrict__ Wfc, float* __restrict__ out) {
    int lane = threadIdx.x & 31;
    int g = blockIdx.x * 8 + (threadIdx.x >> 5);
    if (g >= NGRAPH) return;
    float4 p = *(const float4*)(pool + (size_t)g * HD + lane * 4);
    float4 w = *(const float4*)(Wfc + lane * 4);
    float part = p.x * w.x + p.y * w.y + p.z * w.z + p.w * w.w;
#pragma unroll
    for (int m = 16; m >= 1; m >>= 1) part += __shfl_xor(part, m, 32);
    if (lane == 0) {
        float c = fmaxf(cnt[g], 1.0f);
        out[g] = 1.0f / (1.0f + __expf(-part / c));
    }
}

// ---------------- launch ----------------

extern "C" void kernel_launch(void* const* d_in, const int* in_sizes, int n_in,
                              void* d_out, int out_size, void* d_ws, size_t ws_size,
                              hipStream_t stream) {
    const float*     x     = (const float*)d_in[0];
    const long long* ei    = (const long long*)d_in[1];
    const long long* batch = (const long long*)d_in[2];
    const float* W1 = (const float*)d_in[3];
    const float* g1 = (const float*)d_in[4], *b1 = (const float*)d_in[5];
    const float* m1 = (const float*)d_in[6], *v1 = (const float*)d_in[7];
    const float* W2 = (const float*)d_in[8];
    const float* g2 = (const float*)d_in[9],  *b2 = (const float*)d_in[10];
    const float* m2 = (const float*)d_in[11], *v2 = (const float*)d_in[12];
    const float* W3 = (const float*)d_in[13];
    const float* g3 = (const float*)d_in[14], *b3 = (const float*)d_in[15];
    const float* m3 = (const float*)d_in[16], *v3 = (const float*)d_in[17];
    const float* Wfc = (const float*)d_in[18];

    // Workspace layout (all offsets 16B aligned), total ~104.3 MB
    char* w = (char*)d_ws;
    float*  buf0 = (float*)(w);                        // 51,200,000 B
    float*  buf1 = (float*)(w + 51200000);             // 51,200,000 B
    float*  S4   = (float*)(w + 102400000);            //  1,600,000 B
    __bf16* Wb2  = (__bf16*)(w + 104000000);           //     32,768 B
    __bf16* Wb3  = (__bf16*)(w + 104032768);           //     32,768 B
    float*  ss   = (float*)(w + 104065536);            //      3,072 B (3x scale+shift)
    float*  pool = (float*)(w + 104068608);            //    131,072 B
    float*  cnt  = (float*)(w + 104199680);            //      1,024 B

    // --- prep ---
    pack_w<<<64, 256, 0, stream>>>(W2, Wb2);
    pack_w<<<64, 256, 0, stream>>>(W3, Wb3);
    bn_prep<<<1, 128, 0, stream>>>(g1, b1, m1, v1, ss + 0,   ss + 128);
    bn_prep<<<1, 128, 0, stream>>>(g2, b2, m2, v2, ss + 256, ss + 384);
    bn_prep<<<1, 128, 0, stream>>>(g3, b3, m3, v3, ss + 512, ss + 640);
    zero_f<<<129, 256, 0, stream>>>(pool, NGRAPH * HD + NGRAPH);   // pool + cnt contiguous

    // --- layer 1:  S4 = x ; scatter-add edges ; h1 = relu(bn(S4 @ W1)) -> buf0 ---
    copy4<<<(NNODES + 255) / 256, 256, 0, stream>>>((const float4*)x, (float4*)S4, NNODES);
    edge_aggr4<<<NEDGES / 256, 256, 0, stream>>>(x, ei, S4);
    gin_layer1<<<(NNODES * HD) / 256, 256, 0, stream>>>(S4, W1, ss + 0, ss + 128, buf0);

    // --- layer 2:  S = h1 + aggr(h1) in buf1 ; h2 = relu(bn(S @ W2)) -> buf0 ---
    copy4<<<(NNODES * HD / 4) / 256, 256, 0, stream>>>((const float4*)buf0, (float4*)buf1,
                                                       NNODES * HD / 4);
    edge_aggr128<<<NEDGES / 8, 256, 0, stream>>>(buf0, ei, buf1);
    gin_gemm_wmma<<<(NNODES / 16 + 7) / 8, 256, 0, stream>>>(buf1, Wb2, ss + 256, ss + 384, buf0);

    // --- layer 3 ---
    copy4<<<(NNODES * HD / 4) / 256, 256, 0, stream>>>((const float4*)buf0, (float4*)buf1,
                                                       NNODES * HD / 4);
    edge_aggr128<<<NEDGES / 8, 256, 0, stream>>>(buf0, ei, buf1);
    gin_gemm_wmma<<<(NNODES / 16 + 7) / 8, 256, 0, stream>>>(buf1, Wb3, ss + 512, ss + 640, buf0);

    // --- mean pool per graph + fc + sigmoid ---
    pool_kernel<<<NNODES / 8, 256, 0, stream>>>(buf0, batch, pool, cnt);
    head_kernel<<<NGRAPH / 8, 256, 0, stream>>>(pool, cnt, Wfc, (float*)d_out);
}
